// InformationMaskOptimizationModule_55473797595935
// MI455X (gfx1250) — compile-verified, matrix operand order checked by reading, not run on hardware
//
#include <hip/hip_runtime.h>
#include <math.h>

// ---------------- problem constants ----------------
#define NROWS   32
#define PELEMS  (1u << 21)       // 2,097,152 elements per row
#define EDIM    64
#define HDIM    128
#define NLVL    16
#define NB      4096             // fine histogram bins (256 per entropy level)
#define CHUNKS  64               // blocks per row for streaming passes
#define LN_EPS  1e-5f

typedef __attribute__((ext_vector_type(16))) _Float16 v16h;
typedef __attribute__((ext_vector_type(8)))  float    v8f;

// ---------------- workspace layout (32-bit words) ----------------
#define WS_SUM    0
#define WS_SUMSQ  32
#define WS_GSUM   64
#define WS_GSUMSQ 96
#define WS_UMIN   128
#define WS_UMAX   160
#define WS_HIST   192
#define WS_WORDS  (192 + NROWS * NB)

// ordered-uint mapping so unsigned atomicMin/Max implement float min/max
__device__ __forceinline__ unsigned ford(float f) {
  unsigned u = __float_as_uint(f);
  return (u & 0x80000000u) ? ~u : (u | 0x80000000u);
}
__device__ __forceinline__ float funord(unsigned u) {
  return __uint_as_float((u & 0x80000000u) ? (u ^ 0x80000000u) : ~u);
}

// ---------------- pass 0: workspace init (must run every launch) ----------------
__global__ void k_init(unsigned* __restrict__ ws) {
  int i = blockIdx.x * blockDim.x + threadIdx.x;
  if (i >= WS_WORDS) return;
  unsigned v = 0u;
  if (i >= WS_UMIN && i < WS_UMAX) v = 0xFFFFFFFFu;  // ordered +inf for min
  ws[i] = v;
}

// ---------------- pass 1: moments + min/max (params) + fisher moments (grads^2) ----
__global__ void __launch_bounds__(256) k_reduce(const float* __restrict__ params,
                                                const float* __restrict__ grads,
                                                unsigned* __restrict__ ws) {
  const int row = blockIdx.x;
  const int chunk = blockIdx.y;
  const int tid = threadIdx.x;
  const int n4 = PELEMS / CHUNKS / 4;  // 8192 float4 per chunk
  const float4* p4 =
      (const float4*)(params + (size_t)row * PELEMS) + (size_t)chunk * n4;
  const float4* g4 =
      (const float4*)(grads + (size_t)row * PELEMS) + (size_t)chunk * n4;

  float s = 0.f, ss = 0.f, gs = 0.f, gss = 0.f;
  float mn = INFINITY, mx = -INFINITY;
  for (int i = tid; i < n4; i += 256) {
    __builtin_prefetch(p4 + i + 2048, 0, 1);  // global_prefetch_b8, ~32KB ahead
    __builtin_prefetch(g4 + i + 2048, 0, 1);
    float4 p = p4[i];
    float4 g = g4[i];
    s += p.x + p.y + p.z + p.w;
    ss += p.x * p.x + p.y * p.y + p.z * p.z + p.w * p.w;
    mn = fminf(mn, fminf(fminf(p.x, p.y), fminf(p.z, p.w)));
    mx = fmaxf(mx, fmaxf(fmaxf(p.x, p.y), fmaxf(p.z, p.w)));
    float fx = g.x * g.x, fy = g.y * g.y, fz = g.z * g.z, fw = g.w * g.w;
    gs += fx + fy + fz + fw;
    gss += fx * fx + fy * fy + fz * fz + fw * fw;
  }

  // wave32 shuffle reduction
  for (int o = 16; o > 0; o >>= 1) {
    s += __shfl_down(s, o, 32);
    ss += __shfl_down(ss, o, 32);
    gs += __shfl_down(gs, o, 32);
    gss += __shfl_down(gss, o, 32);
    mn = fminf(mn, __shfl_down(mn, o, 32));
    mx = fmaxf(mx, __shfl_down(mx, o, 32));
  }
  __shared__ float sh[6][8];
  const int w = tid >> 5, l = tid & 31;
  if (l == 0) {
    sh[0][w] = s; sh[1][w] = ss; sh[2][w] = gs;
    sh[3][w] = gss; sh[4][w] = mn; sh[5][w] = mx;
  }
  __syncthreads();
  if (tid == 0) {
    for (int i = 1; i < 8; ++i) {
      s += sh[0][i]; ss += sh[1][i]; gs += sh[2][i]; gss += sh[3][i];
      mn = fminf(mn, sh[4][i]); mx = fmaxf(mx, sh[5][i]);
    }
    float* fws = (float*)ws;
    atomicAdd(fws + WS_SUM + row, s);
    atomicAdd(fws + WS_SUMSQ + row, ss);
    atomicAdd(fws + WS_GSUM + row, gs);
    atomicAdd(fws + WS_GSUMSQ + row, gss);
    atomicMin(ws + WS_UMIN + row, ford(mn));
    atomicMax(ws + WS_UMAX + row, ford(mx));
  }
}

// ---------------- pass 2: fine histogram (LDS atomics, merged globally) ---------
__global__ void __launch_bounds__(256) k_hist(const float* __restrict__ params,
                                              unsigned* __restrict__ ws) {
  const int row = blockIdx.x;
  const int chunk = blockIdx.y;
  const int tid = threadIdx.x;
  __shared__ unsigned lh[NB];
  for (int i = tid; i < NB; i += 256) lh[i] = 0u;
  __syncthreads();

  const float mn = funord(ws[WS_UMIN + row]);
  const float mx = funord(ws[WS_UMAX + row]);
  const float scale = (mx > mn) ? (float)NB / (mx - mn) : 0.0f;

  const int n4 = PELEMS / CHUNKS / 4;
  const float4* p4 =
      (const float4*)(params + (size_t)row * PELEMS) + (size_t)chunk * n4;
  for (int i = tid; i < n4; i += 256) {
    __builtin_prefetch(p4 + i + 2048, 0, 1);
    float4 p = p4[i];
    float v[4] = {p.x, p.y, p.z, p.w};
#pragma unroll
    for (int k = 0; k < 4; ++k) {
      int b = (int)((v[k] - mn) * scale);
      b = min(max(b, 0), NB - 1);
      atomicAdd(&lh[b], 1u);  // ds_add_u32
    }
  }
  __syncthreads();
  for (int i = tid; i < NB; i += 256) {
    unsigned c = lh[i];
    if (c) atomicAdd(ws + WS_HIST + (size_t)row * NB + i, c);
  }
}

// ---------------- MLP head on one wave32 with v_wmma_f32_16x16x32_f16 -----------
__device__ float head_forward(float (*feat)[EDIM], float (*hbuf)[HDIM],
                              const float* __restrict__ w1, const float* __restrict__ b1,
                              const float* __restrict__ g, const float* __restrict__ bt,
                              const float* __restrict__ w2, const float* __restrict__ b2,
                              int lane, bool softplus) {
  const int mloc = lane & 15;
  const int khalf = (lane < 16) ? 0 : 8;  // A/B 16-bit layout: lane half selects K group
  const int rbase = (lane >= 16) ? 8 : 0; // C layout: VGPR r -> row r + 8*(lane>=16)

#pragma unroll
  for (int mt = 0; mt < 2; ++mt) {
    const int mrow = mt * 16 + mloc;
#pragma unroll
    for (int nt = 0; nt < 8; ++nt) {
      const int ncol = nt * 16 + mloc;
      v8f acc = {};
#pragma unroll
      for (int kt = 0; kt < 2; ++kt) {
        const int kb = kt * 32 + khalf;
        v16h a, bm;
#pragma unroll
        for (int e = 0; e < 8; ++e) {
          a[e]      = (_Float16)feat[mrow][kb + e];
          a[e + 8]  = (_Float16)feat[mrow][kb + 16 + e];
          bm[e]     = (_Float16)w1[(kb + e) * HDIM + ncol];
          bm[e + 8] = (_Float16)w1[(kb + 16 + e) * HDIM + ncol];
        }
        acc = __builtin_amdgcn_wmma_f32_16x16x32_f16(
            false, a, false, bm, (short)0, acc, false, false);
      }
#pragma unroll
      for (int r = 0; r < 8; ++r)
        hbuf[mt * 16 + rbase + r][ncol] = acc[r] + b1[ncol];
    }
  }
  __syncthreads();

  // LayerNorm (biased var) + exact GELU + GEMV, one row per lane
  float mu = 0.f;
  for (int k = 0; k < HDIM; ++k) mu += hbuf[lane][k];
  mu *= (1.0f / HDIM);
  float var = 0.f;
  for (int k = 0; k < HDIM; ++k) {
    float d = hbuf[lane][k] - mu;
    var += d * d;
  }
  var *= (1.0f / HDIM);
  const float inv = rsqrtf(var + LN_EPS);
  float acc = 0.f;
  for (int k = 0; k < HDIM; ++k) {
    float hv = g[k] * (hbuf[lane][k] - mu) * inv + bt[k];
    hv = 0.5f * hv * (1.0f + erff(hv * 0.70710678118654752f));  // exact GELU
    acc += hv * w2[k];
  }
  acc += b2[0];
  if (softplus) {
    acc = (acc > 20.0f) ? acc : log1pf(expf(acc));
  } else {
    acc = 1.0f / (1.0f + expf(-acc));
  }
  return acc;
}

// ---------------- pass 3: features + quantiles + entropy + heads ----------------
__global__ void __launch_bounds__(32) k_final(
    const unsigned* __restrict__ ws,
    const float* __restrict__ d_w1, const float* __restrict__ d_b1,
    const float* __restrict__ d_g, const float* __restrict__ d_bt,
    const float* __restrict__ d_w2, const float* __restrict__ d_b2,
    const float* __restrict__ f_w1, const float* __restrict__ f_b1,
    const float* __restrict__ f_g, const float* __restrict__ f_bt,
    const float* __restrict__ f_w2, const float* __restrict__ f_b2,
    float* __restrict__ out) {
  __shared__ float feat[NROWS][EDIM];
  __shared__ float hbuf[NROWS][HDIM];
  const int t = threadIdx.x;  // lane == row, 32 lanes, EXEC all ones
  const float* fws = (const float*)ws;
  const float Pn = (float)PELEMS;

  const float sum = fws[WS_SUM + t], sumsq = fws[WS_SUMSQ + t];
  const float gsum = fws[WS_GSUM + t], gsumsq = fws[WS_GSUMSQ + t];
  const float mn = funord(ws[WS_UMIN + t]);
  const float mx = funord(ws[WS_UMAX + t]);

  const float mean = sum / Pn;
  const float stdv = sqrtf(fmaxf((sumsq - Pn * mean * mean) / (Pn - 1.0f), 0.0f));
  const float fm = gsum / Pn;
  const float fs = sqrtf(fmaxf((gsumsq - Pn * fm * fm) / (Pn - 1.0f), 0.0f));

  // quantiles (rank interpolation in fine bin) + entropy (256 fine bins / level)
  float q10 = mn, q50 = mn, q90 = mn;
  const float t10 = 0.1f * (Pn - 1.0f), t50 = 0.5f * (Pn - 1.0f),
              t90 = 0.9f * (Pn - 1.0f);
  bool f10 = false, f50 = false, f90 = false;
  const float binw = (mx > mn) ? (mx - mn) / (float)NB : 0.0f;
  float cum = 0.0f, entacc = 0.0f;
  unsigned gcnt = 0;
  const unsigned* h = ws + WS_HIST + (size_t)t * NB;
  for (int b = 0; b < NB; ++b) {
    const unsigned c = h[b];
    const float cf = (float)c;
    if (c) {
      if (!f10 && cum + cf > t10) { q10 = mn + ((float)b + (t10 - cum + 0.5f) / cf) * binw; f10 = true; }
      if (!f50 && cum + cf > t50) { q50 = mn + ((float)b + (t50 - cum + 0.5f) / cf) * binw; f50 = true; }
      if (!f90 && cum + cf > t90) { q90 = mn + ((float)b + (t90 - cum + 0.5f) / cf) * binw; f90 = true; }
    }
    cum += cf;
    gcnt += c;
    if ((b & 255) == 255) {  // close one of the 16 entropy levels
      if (gcnt) {
        const float p = (float)gcnt / Pn;
        entacc += p * log2f(p);
      }
      gcnt = 0;
    }
  }
  const float ent = (mx > mn) ? (-entacc * 0.25f) : 0.0f;  // / log2(16)

  // density features: [base(7), ent, 0...]
  for (int k = 0; k < EDIM; ++k) feat[t][k] = 0.0f;
  feat[t][0] = mean; feat[t][1] = stdv; feat[t][2] = mx; feat[t][3] = mn;
  feat[t][4] = q10; feat[t][5] = q50; feat[t][6] = q90;
  feat[t][7] = ent;
  __syncthreads();
  const float density =
      head_forward(feat, hbuf, d_w1, d_b1, d_g, d_bt, d_w2, d_b2, t, false);
  __syncthreads();

  // fisher features: [base(7), 0..., fm, fs]
  feat[t][7] = 0.0f;
  feat[t][EDIM - 2] = fm;
  feat[t][EDIM - 1] = fs;
  __syncthreads();
  const float fisher =
      head_forward(feat, hbuf, f_w1, f_b1, f_g, f_bt, f_w2, f_b2, t, true);

  out[t * 3 + 0] = density;
  out[t * 3 + 1] = fisher;
  out[t * 3 + 2] = ent;
}

extern "C" void kernel_launch(void* const* d_in, const int* in_sizes, int n_in,
                              void* d_out, int out_size, void* d_ws, size_t ws_size,
                              hipStream_t stream) {
  const float* params = (const float*)d_in[0];
  const float* grads = (const float*)d_in[1];
  unsigned* ws = (unsigned*)d_ws;

  k_init<<<(WS_WORDS + 255) / 256, 256, 0, stream>>>(ws);

  dim3 grid(NROWS, CHUNKS);
  k_reduce<<<grid, 256, 0, stream>>>(params, grads, ws);
  k_hist<<<grid, 256, 0, stream>>>(params, ws);

  k_final<<<1, 32, 0, stream>>>(
      ws, (const float*)d_in[2], (const float*)d_in[3], (const float*)d_in[4],
      (const float*)d_in[5], (const float*)d_in[6], (const float*)d_in[7],
      (const float*)d_in[8], (const float*)d_in[9], (const float*)d_in[10],
      (const float*)d_in[11], (const float*)d_in[12], (const float*)d_in[13],
      (float*)d_out);
}
